// Transformer_59596966199565
// MI455X (gfx1250) — compile-verified, matrix operand order checked by reading, not run on hardware
//
#include <hip/hip_runtime.h>
#include <hip/hip_bf16.h>

#define Bv   8
#define Nv   256
#define Mv   256
#define Hv   1024
#define NSv  16
#define Lv   6
#define FFv  4096
#define HDv  64
#define VOUTv 32000

typedef __attribute__((ext_vector_type(16))) __bf16 v16bf;
typedef __attribute__((ext_vector_type(8)))  float  v8f;

// Load a 16-element bf16 fragment as two 16-byte LDS reads (ds_load_b128).
__device__ __forceinline__ v16bf load_frag(const __bf16* p0, const __bf16* p1) {
  union { uint4 q[2]; v16bf v; } u;
  u.q[0] = *(const uint4*)p0;
  u.q[1] = *(const uint4*)p1;
  return u.v;
}

// ---------------------------------------------------------------------------
// Embedding lookup + sinusoidal positional encoding (computed on the fly)
// ---------------------------------------------------------------------------
__global__ void embed_pe_kernel(const float* __restrict__ table,
                                const int* __restrict__ tok,
                                float* __restrict__ out, int T) {
  int i = blockIdx.x * blockDim.x + threadIdx.x;
  int total = Bv * T * Hv;
  if (i >= total) return;
  int c = i & (Hv - 1);
  int t = (i >> 10) % T;           // Hv == 1024
  int b = i / (T * Hv);
  int tk = tok[b * T + t];
  float dim = (float)(c >> 1);
  float ang = (float)t * __expf(-dim * (9.210340371976184f / (float)Hv)); // ln(10000)
  float pe = (c & 1) ? __cosf(ang) : __sinf(ang);
  out[i] = table[(size_t)tk * Hv + c] + pe;
}

// ---------------------------------------------------------------------------
// LayerNorm: one block (256 threads) per row of H=1024
// ---------------------------------------------------------------------------
__global__ void __launch_bounds__(256)
layernorm_kernel(const float* __restrict__ x, const float* __restrict__ g,
                 const float* __restrict__ bt, float* __restrict__ y) {
  __shared__ float red[256];
  int tid = threadIdx.x;
  const float* xr = x + (size_t)blockIdx.x * Hv;
  float s = 0.f;
  for (int c = tid; c < Hv; c += 256) s += xr[c];
  red[tid] = s; __syncthreads();
  for (int st = 128; st > 0; st >>= 1) {
    if (tid < st) red[tid] += red[tid + st];
    __syncthreads();
  }
  float mu = red[0] * (1.0f / Hv);
  __syncthreads();
  float v = 0.f;
  for (int c = tid; c < Hv; c += 256) { float d = xr[c] - mu; v += d * d; }
  red[tid] = v; __syncthreads();
  for (int st = 128; st > 0; st >>= 1) {
    if (tid < st) red[tid] += red[tid + st];
    __syncthreads();
  }
  float inv = rsqrtf(red[0] * (1.0f / Hv) + 1e-5f);
  float* yr = y + (size_t)blockIdx.x * Hv;
  for (int c = tid; c < Hv; c += 256) yr[c] = (xr[c] - mu) * inv * g[c] + bt[c];
}

// ---------------------------------------------------------------------------
// Tiled bf16 WMMA GEMM: C = act(A@B + bias) (+ residual)
// A[M,K] f32 row-major, B[K,N] f32 row-major, C[M,N] f32.
// Block 256 thr = 8 waves; tile 128x64x32; wave tile 32x32 (4 WMMAs/K-step).
// B tile staged TRANSPOSED in LDS so both operands' fragments are two
// contiguous 16B runs per lane (ds_load_b128 x2 per fragment).
// Assumes M%128==0, N%64==0, K%32==0 (true for all shapes in this model).
// ---------------------------------------------------------------------------
#define APITCH 40   // 80B rows: 16B-aligned, decent bank spread
#define BPITCH 40

template<bool HAS_BIAS, bool RELU, bool HAS_RES>
__global__ void __launch_bounds__(256)
gemm_bf16_kernel(const float* __restrict__ A, const float* __restrict__ Bm,
                 const float* __restrict__ bias, const float* __restrict__ res,
                 float* __restrict__ C, int Mdim, int Ndim, int Kdim) {
  __shared__ __align__(16) __bf16 As[128][APITCH];   // [m][k]
  __shared__ __align__(16) __bf16 Bt[64][BPITCH];    // [n][k]  (transposed)
  const int tid  = threadIdx.x;
  const int lane = tid & 31;
  const int w    = tid >> 5;
  const int wm   = w >> 1, wn = w & 1;     // 4x2 wave grid over 128x64 tile
  const int m0   = blockIdx.y * 128, n0 = blockIdx.x * 64;
  const int ar   = tid >> 1, ac = (tid & 1) << 4;   // A loader: 128 rows x 2 halves
  const int br   = tid >> 3, bc = (tid & 7) << 3;   // B loader: 32 k-rows x 8 n-groups
  // wave32 WMMA striping (ISA 7.12.2)
  const int mr = lane & 15;
  const int kb = (lane >> 4) << 3;
  const int nc = lane & 15;
  const int mb = (lane >> 4) << 3;

  v8f acc[2][2];
  v8f z8 = {0,0,0,0,0,0,0,0};
  acc[0][0] = z8; acc[0][1] = z8; acc[1][0] = z8; acc[1][1] = z8;

  for (int k0 = 0; k0 < Kdim; k0 += 32) {
    const float* Ap = A  + (size_t)(m0 + ar) * Kdim + k0 + ac;
    const float* Bp = Bm + (size_t)(k0 + br) * Ndim + n0 + bc;
    // A: 16 consecutive f32 -> 16 bf16 -> two 16B LDS stores
    {
      union { uint4 q; __bf16 h[8]; } p0, p1;
#pragma unroll
      for (int j = 0; j < 8; ++j) { p0.h[j] = (__bf16)Ap[j]; p1.h[j] = (__bf16)Ap[8 + j]; }
      *(uint4*)&As[ar][ac]     = p0.q;
      *(uint4*)&As[ar][ac + 8] = p1.q;
    }
    // B: 8 consecutive f32 of one k-row -> transpose-scatter into Bt[n][k]
#pragma unroll
    for (int j = 0; j < 8; ++j) Bt[bc + j][br] = (__bf16)Bp[j];
    if (k0 + 32 < Kdim) {                       // gfx1250 global_prefetch_b8
      __builtin_prefetch(Ap + 32, 0, 1);
      __builtin_prefetch(Bp + (size_t)32 * Ndim, 0, 1);
    }
    __syncthreads();

    v16bf a0 = load_frag(&As[wm * 32 + mr][kb],      &As[wm * 32 + mr][kb + 16]);
    v16bf a1 = load_frag(&As[wm * 32 + 16 + mr][kb], &As[wm * 32 + 16 + mr][kb + 16]);
    v16bf b0 = load_frag(&Bt[wn * 32 + nc][kb],      &Bt[wn * 32 + nc][kb + 16]);
    v16bf b1 = load_frag(&Bt[wn * 32 + 16 + nc][kb], &Bt[wn * 32 + 16 + nc][kb + 16]);
    acc[0][0] = __builtin_amdgcn_wmma_f32_16x16x32_bf16(false, a0, false, b0, (short)0, acc[0][0], false, false);
    acc[0][1] = __builtin_amdgcn_wmma_f32_16x16x32_bf16(false, a0, false, b1, (short)0, acc[0][1], false, false);
    acc[1][0] = __builtin_amdgcn_wmma_f32_16x16x32_bf16(false, a1, false, b0, (short)0, acc[1][0], false, false);
    acc[1][1] = __builtin_amdgcn_wmma_f32_16x16x32_bf16(false, a1, false, b1, (short)0, acc[1][1], false, false);
    __syncthreads();
  }
#pragma unroll
  for (int ri = 0; ri < 2; ++ri)
#pragma unroll
    for (int ci = 0; ci < 2; ++ci)
#pragma unroll
      for (int r = 0; r < 8; ++r) {
        int m = m0 + wm * 32 + ri * 16 + mb + r;
        int c = n0 + wn * 32 + ci * 16 + nc;
        float v = acc[ri][ci][r];
        if (HAS_BIAS) v += bias[c];
        if (RELU)     v = fmaxf(v, 0.f);
        size_t idx = (size_t)m * Ndim + c;
        if (HAS_RES)  v += res[idx];
        C[idx] = v;
      }
}

// ---------------------------------------------------------------------------
// Fused attention: one wave per (16 query rows, head, batch).
// Q,K,V are the projected [B, seq, H] tensors; head h uses channels h*64..h*64+63.
// MASKMODE 0: pad mask (key >= x_length[b]); MASKMODE 1: causal (key > query).
// scores = (Q*scale)@K^T -> mask -> softmax -> @V, all WMMA bf16.
// V tile staged transposed so its fragments are contiguous too.
// ---------------------------------------------------------------------------
#define QPITCH 72    // 144B rows: 16B aligned, conflict-free 16-lane stride
#define PPITCH 264   // probs pitch (bf16): multiple of 8, avoids 512B-stride conflicts

template<int MASKMODE>
__global__ void __launch_bounds__(32)
attention_kernel(const float* __restrict__ Qp, const float* __restrict__ Kp,
                 const float* __restrict__ Vp, const int* __restrict__ xlen,
                 float* __restrict__ Out, int NQ, int NK) {
  __shared__ float  sc[16][257];
  __shared__ __align__(16) __bf16 pb[16][PPITCH];
  __shared__ __align__(16) __bf16 ql[16][QPITCH];
  __shared__ __align__(16) __bf16 kl[16][QPITCH];
  __shared__ __align__(16) __bf16 vlT[64][BPITCH];   // [hd][key] transposed
  const int lane = threadIdx.x;
  const int q0 = blockIdx.x * 16;
  const int h  = blockIdx.y;
  const int b  = blockIdx.z;
  const int lim = (MASKMODE == 0) ? xlen[b] : 0;
  const float scale = 0.125f;  // 1/sqrt(HD=64)

  for (int i = lane; i < 16 * 64; i += 32) {
    int r = i >> 6, c = i & 63;
    ql[r][c] = (__bf16)(Qp[(size_t)(b * NQ + q0 + r) * Hv + h * HDv + c] * scale);
  }
  __syncthreads();

  const int mr = lane & 15, kb = (lane >> 4) << 3;
  const int nc = lane & 15, mb = (lane >> 4) << 3;
  v16bf a0 = load_frag(&ql[mr][kb],      &ql[mr][kb + 16]);
  v16bf a1 = load_frag(&ql[mr][32 + kb], &ql[mr][32 + kb + 16]);

  // ---- scores: Q @ K^T over HD=64 (two WMMAs per 16-key tile) ----
  for (int kt = 0; kt < NK; kt += 16) {
    for (int i = lane; i < 16 * 64; i += 32) {
      int r = i >> 6, c = i & 63;
      kl[r][c] = (__bf16)Kp[(size_t)(b * NK + kt + r) * Hv + h * HDv + c];
    }
    __syncthreads();
    v16bf b0 = load_frag(&kl[nc][kb],      &kl[nc][kb + 16]);
    v16bf b1 = load_frag(&kl[nc][32 + kb], &kl[nc][32 + kb + 16]);
    v8f s = {0,0,0,0,0,0,0,0};
    s = __builtin_amdgcn_wmma_f32_16x16x32_bf16(false, a0, false, b0, (short)0, s, false, false);
    s = __builtin_amdgcn_wmma_f32_16x16x32_bf16(false, a1, false, b1, (short)0, s, false, false);
    int key = kt + nc;
#pragma unroll
    for (int r = 0; r < 8; ++r) {
      int m = mb + r;
      bool masked = (MASKMODE == 0) ? (key >= lim) : (key > q0 + m);
      sc[m][key] = masked ? -__builtin_inff() : s[r];
    }
    __syncthreads();
  }

  // ---- row softmax (lanes 0..15, one row each) ----
  if (lane < 16) {
    float mx = -__builtin_inff();
    for (int j = 0; j < NK; ++j) mx = fmaxf(mx, sc[lane][j]);
    float sum = 0.f;
    for (int j = 0; j < NK; ++j) { float e = __expf(sc[lane][j] - mx); sum += e; sc[lane][j] = e; }
    float inv = 1.f / sum;
    for (int j = 0; j < NK; ++j) pb[lane][j] = (__bf16)(sc[lane][j] * inv);
  }
  __syncthreads();

  // ---- out = P @ V : 4 output tiles of 16 cols, K-chunks of 32 keys ----
  v8f z8 = {0,0,0,0,0,0,0,0};
  v8f ot[4];
#pragma unroll
  for (int t = 0; t < 4; ++t) ot[t] = z8;

  for (int kc = 0; kc < NK; kc += 32) {
    for (int i = lane; i < 32 * 64; i += 32) {
      int r = i >> 6, c = i & 63;      // r = key-in-chunk, c = hd channel
      vlT[c][r] = (__bf16)Vp[(size_t)(b * NK + kc + r) * Hv + h * HDv + c];
    }
    __syncthreads();
    v16bf pfrag = load_frag(&pb[mr][kc + kb], &pb[mr][kc + kb + 16]);
#pragma unroll
    for (int t = 0; t < 4; ++t) {
      v16bf bv = load_frag(&vlT[t * 16 + nc][kb], &vlT[t * 16 + nc][kb + 16]);
      ot[t] = __builtin_amdgcn_wmma_f32_16x16x32_bf16(false, pfrag, false, bv, (short)0, ot[t], false, false);
    }
    __syncthreads();
  }
#pragma unroll
  for (int t = 0; t < 4; ++t)
#pragma unroll
    for (int r = 0; r < 8; ++r) {
      int m = mb + r;
      Out[(size_t)(b * NQ + q0 + m) * Hv + h * HDv + t * 16 + nc] = ot[t][r];
    }
}

// ---------------------------------------------------------------------------
// In-place log-softmax over VOUT=32000 per row
// ---------------------------------------------------------------------------
__global__ void __launch_bounds__(256)
logsoftmax_kernel(float* __restrict__ x) {
  __shared__ float red[256];
  int tid = threadIdx.x;
  float* xr = x + (size_t)blockIdx.x * VOUTv;
  float mx = -__builtin_inff();
  for (int j = tid; j < VOUTv; j += 256) mx = fmaxf(mx, xr[j]);
  red[tid] = mx; __syncthreads();
  for (int st = 128; st > 0; st >>= 1) {
    if (tid < st) red[tid] = fmaxf(red[tid], red[tid + st]);
    __syncthreads();
  }
  mx = red[0]; __syncthreads();
  float s = 0.f;
  for (int j = tid; j < VOUTv; j += 256) s += __expf(xr[j] - mx);
  red[tid] = s; __syncthreads();
  for (int st = 128; st > 0; st >>= 1) {
    if (tid < st) red[tid] += red[tid + st];
    __syncthreads();
  }
  float lse = mx + __logf(red[0]);
  for (int j = tid; j < VOUTv; j += 256) xr[j] -= lse;
}

// ---------------------------------------------------------------------------
// Host orchestration
// ---------------------------------------------------------------------------
static inline void gemm_launch(hipStream_t st, const float* A, const float* B,
                               const float* bias, const float* res, float* C,
                               int M, int N, int K, bool relu) {
  dim3 g(N / 64, M / 128), blk(256);
  if (bias && res)
    gemm_bf16_kernel<true, false, true><<<g, blk, 0, st>>>(A, B, bias, res, C, M, N, K);
  else if (bias && relu)
    gemm_bf16_kernel<true, true, false><<<g, blk, 0, st>>>(A, B, bias, nullptr, C, M, N, K);
  else if (bias)
    gemm_bf16_kernel<true, false, false><<<g, blk, 0, st>>>(A, B, bias, nullptr, C, M, N, K);
  else if (res)
    gemm_bf16_kernel<false, false, true><<<g, blk, 0, st>>>(A, B, nullptr, res, C, M, N, K);
  else
    gemm_bf16_kernel<false, false, false><<<g, blk, 0, st>>>(A, B, nullptr, nullptr, C, M, N, K);
}

extern "C" void kernel_launch(void* const* d_in, const int* in_sizes, int n_in,
                              void* d_out, int out_size, void* d_ws, size_t ws_size,
                              hipStream_t stream) {
  (void)in_sizes; (void)n_in; (void)out_size; (void)ws_size;
  // --- inputs, setup_inputs() dict order (params flattened recursively) ---
  const float* embed_enc = (const float*)d_in[0];
  const float* embed_dec = (const float*)d_in[1];
  const float* e_ln1_g = (const float*)d_in[2];
  const float* e_ln1_b = (const float*)d_in[3];
  const float* e_ln2_g = (const float*)d_in[4];
  const float* e_ln2_b = (const float*)d_in[5];
  const float* e_Wq = (const float*)d_in[6];
  const float* e_Wk = (const float*)d_in[7];
  const float* e_Wv = (const float*)d_in[8];
  const float* e_Wo = (const float*)d_in[9];
  const float* e_W1 = (const float*)d_in[10];
  const float* e_b1 = (const float*)d_in[11];
  const float* e_W2 = (const float*)d_in[12];
  const float* e_b2 = (const float*)d_in[13];
  const float* d_lnm_g = (const float*)d_in[14];
  const float* d_lnm_b = (const float*)d_in[15];
  const float* d_lna_g = (const float*)d_in[16];
  const float* d_lna_b = (const float*)d_in[17];
  const float* d_lnf_g = (const float*)d_in[18];
  const float* d_lnf_b = (const float*)d_in[19];
  const float* d_sWq = (const float*)d_in[20];
  const float* d_sWk = (const float*)d_in[21];
  const float* d_sWv = (const float*)d_in[22];
  const float* d_sWo = (const float*)d_in[23];
  const float* d_cWq = (const float*)d_in[24];
  const float* d_cWk = (const float*)d_in[25];
  const float* d_cWv = (const float*)d_in[26];
  const float* d_cWo = (const float*)d_in[27];
  const float* d_W1 = (const float*)d_in[28];
  const float* d_b1 = (const float*)d_in[29];
  const float* d_W2 = (const float*)d_in[30];
  const float* d_b2 = (const float*)d_in[31];
  const float* g_ln_g = (const float*)d_in[32];
  const float* g_ln_b = (const float*)d_in[33];
  const float* g_W = (const float*)d_in[34];
  const float* g_b = (const float*)d_in[35];
  const int* x_tokens = (const int*)d_in[36];
  const int* x_length = (const int*)d_in[37];
  const int* y_tok    = (const int*)d_in[38];

  // --- workspace layout (f32) ---
  float* ws = (float*)d_ws;
  size_t off = 0;
  auto alloc = [&](size_t n) { float* p = ws + off; off += n; return p; };
  const size_t TOK = (size_t)Bv * Nv;  // 2048 (N == M)
  float* z   = alloc(TOK * Hv);
  float* dz  = alloc(TOK * Hv);
  float* lnA = alloc(TOK * Hv);
  float* lnB = alloc(TOK * Hv);
  float* qb  = alloc(TOK * Hv);
  float* kb  = alloc(TOK * Hv);
  float* vb  = alloc(TOK * Hv);
  float* ab  = alloc(TOK * Hv);
  float* ffh = alloc(TOK * FFv);

  const int rows = (int)TOK;
  const size_t HH = (size_t)Hv * Hv, HFF = (size_t)Hv * FFv;

  // --- embeddings + positional encoding ---
  embed_pe_kernel<<<(Bv * Nv * Hv) / 256, 256, 0, stream>>>(embed_enc, x_tokens, z, Nv);
  embed_pe_kernel<<<(Bv * Mv * Hv) / 256, 256, 0, stream>>>(embed_dec, y_tok, dz, Mv);

  dim3 attgE(Nv / 16, NSv, Bv), attgD(Mv / 16, NSv, Bv), attb(32);

  // --- encoder ---
  for (int i = 0; i < Lv; ++i) {
    layernorm_kernel<<<rows, 256, 0, stream>>>(z, e_ln1_g + i * Hv, e_ln1_b + i * Hv, lnA);
    gemm_launch(stream, lnA, e_Wq + i * HH, nullptr, nullptr, qb, rows, Hv, Hv, false);
    gemm_launch(stream, lnA, e_Wk + i * HH, nullptr, nullptr, kb, rows, Hv, Hv, false);
    gemm_launch(stream, lnA, e_Wv + i * HH, nullptr, nullptr, vb, rows, Hv, Hv, false);
    attention_kernel<0><<<attgE, attb, 0, stream>>>(qb, kb, vb, x_length, ab, Nv, Nv);
    gemm_launch(stream, ab, e_Wo + i * HH, nullptr, z, z, rows, Hv, Hv, false);
    layernorm_kernel<<<rows, 256, 0, stream>>>(z, e_ln2_g + i * Hv, e_ln2_b + i * Hv, lnA);
    gemm_launch(stream, lnA, e_W1 + i * HFF, e_b1 + i * FFv, nullptr, ffh, rows, FFv, Hv, true);
    gemm_launch(stream, ffh, e_W2 + i * HFF, e_b2 + i * Hv, z, z, rows, Hv, FFv, false);
  }

  // --- decoder ---
  for (int i = 0; i < Lv; ++i) {
    // masked self-attention
    layernorm_kernel<<<rows, 256, 0, stream>>>(dz, d_lnm_g + i * Hv, d_lnm_b + i * Hv, lnA);
    gemm_launch(stream, lnA, d_sWq + i * HH, nullptr, nullptr, qb, rows, Hv, Hv, false);
    gemm_launch(stream, lnA, d_sWk + i * HH, nullptr, nullptr, kb, rows, Hv, Hv, false);
    gemm_launch(stream, lnA, d_sWv + i * HH, nullptr, nullptr, vb, rows, Hv, Hv, false);
    attention_kernel<1><<<attgD, attb, 0, stream>>>(qb, kb, vb, x_length, ab, Mv, Mv);
    gemm_launch(stream, ab, d_sWo + i * HH, nullptr, dz, dz, rows, Hv, Hv, false);
    // cross-attention (shared lna LN applied to both memory and query)
    layernorm_kernel<<<rows, 256, 0, stream>>>(z,  d_lna_g + i * Hv, d_lna_b + i * Hv, lnA); // kvn
    layernorm_kernel<<<rows, 256, 0, stream>>>(dz, d_lna_g + i * Hv, d_lna_b + i * Hv, lnB); // qn
    gemm_launch(stream, lnB, d_cWq + i * HH, nullptr, nullptr, qb, rows, Hv, Hv, false);
    gemm_launch(stream, lnA, d_cWk + i * HH, nullptr, nullptr, kb, rows, Hv, Hv, false);
    gemm_launch(stream, lnA, d_cWv + i * HH, nullptr, nullptr, vb, rows, Hv, Hv, false);
    attention_kernel<0><<<attgD, attb, 0, stream>>>(qb, kb, vb, x_length, ab, Mv, Nv);
    gemm_launch(stream, ab, d_cWo + i * HH, nullptr, dz, dz, rows, Hv, Hv, false);
    // FFN
    layernorm_kernel<<<rows, 256, 0, stream>>>(dz, d_lnf_g + i * Hv, d_lnf_b + i * Hv, lnA);
    gemm_launch(stream, lnA, d_W1 + i * HFF, d_b1 + i * FFv, nullptr, ffh, rows, FFv, Hv, true);
    gemm_launch(stream, ffh, d_W2 + i * HFF, d_b2 + i * Hv, dz, dz, rows, Hv, FFv, false);
  }

  // --- generator: LN -> logits (into d_out) -> in-place log_softmax ---
  layernorm_kernel<<<rows, 256, 0, stream>>>(dz, g_ln_g, g_ln_b, lnA);
  gemm_launch(stream, lnA, g_W, g_b, nullptr, (float*)d_out, rows, VOUTv, Hv, false);
  logsoftmax_kernel<<<rows, 256, 0, stream>>>((float*)d_out);
}
